// PointNetMSGRandomInput_32375463477353
// MI455X (gfx1250) — matrix operand order repaired and checked by reading, not compiled, not run on hardware
//
#include <hip/hip_runtime.h>
#include <hip/hip_bf16.h>

// ---------------------------------------------------------------------------
// PointNet++ MSG set-abstraction for MI455X (gfx1250, wave32, WMMA f16->f32)
// BN statistics fused into the GEMM epilogue (cross-lane butterfly + atomics)
// ---------------------------------------------------------------------------

#define B_   8
#define N_   4096
#define S_   1024
#define Q_   (B_ * S_)      // 8192 query points
#define INCH 64
#define EPS_ 1e-5f

typedef __attribute__((ext_vector_type(16))) _Float16 v16h;
typedef __attribute__((ext_vector_type(8)))  _Float16 v8h;
typedef __attribute__((ext_vector_type(8)))  float    v8f;

// ------------------------- workspace layout (bytes) ------------------------
static const unsigned long long OFF_B  = 100663296ull;            // bufA size
static const unsigned long long OFF_XF = OFF_B  + 134217728ull;   // bufB end
static const unsigned long long OFF_WP = OFF_XF + 5242880ull;     // xfin end
static const unsigned long long OFF_ST = OFF_WP + 393216ull;      // wp end

// ----------------------------- grouping --------------------------------

__device__ __forceinline__ void write_point(_Float16* __restrict__ Xg,
                                            int q, int K, int slot, int b, int j,
                                            float qx, float qy, float qz,
                                            const float* __restrict__ feat,
                                            const float* __restrict__ xyz)
{
    _Float16* dst = Xg + ((size_t)q * K + slot) * 96;   // Cpad = 96 always
    const float* f = feat + ((size_t)b * N_ + j) * INCH;
#pragma unroll 8
    for (int c = 0; c < INCH; ++c) dst[c] = (_Float16)f[c];
    const float* p = xyz + ((size_t)b * N_ + j) * 3;
    dst[64] = (_Float16)(p[0] - qx);
    dst[65] = (_Float16)(p[1] - qy);
    dst[66] = (_Float16)(p[2] - qz);
#pragma unroll
    for (int c = 67; c < 96; ++c) dst[c] = (_Float16)0.f;
}

// One wave32 per query point. Ordered-index first-K ball query via ballot,
// replicating the reference semantics (sorted idx, pad with first neighbor).
__global__ __launch_bounds__(256)
void group_kernel(const float* __restrict__ new_xyz,
                  const float* __restrict__ xyz,
                  const float* __restrict__ feat,
                  _Float16* __restrict__ Xg,
                  float r2, int K)
{
    const int lane = threadIdx.x & 31;
    const int q    = blockIdx.x * 8 + (threadIdx.x >> 5);
    if (q >= Q_) return;
    const int b = q / S_;
    const float qx = new_xyz[(size_t)q * 3 + 0];
    const float qy = new_xyz[(size_t)q * 3 + 1];
    const float qz = new_xyz[(size_t)q * 3 + 2];

    int cnt = 0, first_j = -1;
    for (int jb = 0; jb < N_ && cnt < K; jb += 32) {
        const int j = jb + lane;
        const float* p = xyz + ((size_t)b * N_ + j) * 3;
        const float dx = p[0] - qx, dy = p[1] - qy, dz = p[2] - qz;
        const float d2 = dx * dx + dy * dy + dz * dz;
        const bool pred = (d2 <= r2);
        const unsigned mask = (unsigned)__ballot(pred);
        if (first_j < 0 && mask) first_j = jb + (__ffs(mask) - 1);
        const int slot = cnt + __popc(mask & ((1u << lane) - 1u));
        if (pred && slot < K)
            write_point(Xg, q, K, slot, b, j, qx, qy, qz, feat, xyz);
        cnt += __popc(mask);
    }
    if (first_j < 0) first_j = 0;   // unreachable (query point is its own neighbor)
    for (int slot = cnt + lane; slot < K; slot += 32)
        write_point(Xg, q, K, slot, b, first_j, qx, qy, qz, feat, xyz);
}

// ----------------------------- WMMA GEMM -----------------------------------
// Y[col*O + o] = sum_c Wp[o*Cpad + c] * X[col*Cpad + c] + bias[o]
// A/B fragments use contiguous k = 16*half + s; BN sum/sumsq accumulated in the epilogue:
// each wave reduces its 16(o) x 64(col) tile across lanes (shfl_xor butterfly
// over the 16 lanes sharing a channel) and issues 2 atomics per channel.
__global__ __launch_bounds__(128)
void wmma_gemm_kernel(const _Float16* __restrict__ Wp,
                      const float*    __restrict__ bias,
                      const _Float16* __restrict__ X,
                      _Float16*       __restrict__ Y,
                      float*          __restrict__ stats,
                      int Cpad, int O, long long cols)
{
    const int lane = threadIdx.x & 31;
    const int wave = threadIdx.x >> 5;
    const int m    = lane & 15;
    const int half = lane >> 4;
    const long long colbase = ((long long)blockIdx.x * 4 + wave) * 64;
    const int o0 = blockIdx.y * 16;

    v8f acc0 = {}, acc1 = {}, acc2 = {}, acc3 = {};

    const _Float16* wrow = Wp + (size_t)(o0 + m) * Cpad;
    const int n  = lane & 15;
    const int kh = half * 16;

    for (int kb = 0; kb < Cpad; kb += 32) {
        const _Float16* pa = wrow + kb + half * 16;
        v16h afrag = *(const v16h*)(pa);

        const _Float16* pb = X + kb + kh;
        v16h b0 = *(const v16h*)(pb + (size_t)(colbase +  0 + n) * Cpad);
        v16h b1 = *(const v16h*)(pb + (size_t)(colbase + 16 + n) * Cpad);
        v16h b2 = *(const v16h*)(pb + (size_t)(colbase + 32 + n) * Cpad);
        v16h b3 = *(const v16h*)(pb + (size_t)(colbase + 48 + n) * Cpad);

        acc0 = __builtin_amdgcn_wmma_f32_16x16x32_f16(false, afrag, false, b0, (short)0, acc0, false, false);
        acc1 = __builtin_amdgcn_wmma_f32_16x16x32_f16(false, afrag, false, b1, (short)0, acc1, false, false);
        acc2 = __builtin_amdgcn_wmma_f32_16x16x32_f16(false, afrag, false, b2, (short)0, acc2, false, false);
        acc3 = __builtin_amdgcn_wmma_f32_16x16x32_f16(false, afrag, false, b3, (short)0, acc3, false, false);
    }

    float bv[8];
#pragma unroll
    for (int r = 0; r < 8; ++r) bv[r] = bias[o0 + 8 * half + r];

    // store (f16) + accumulate per-channel partial sums for BN
    v8f accs[4] = {acc0, acc1, acc2, acc3};
    float ps1[8], ps2[8];
#pragma unroll
    for (int r = 0; r < 8; ++r) { ps1[r] = 0.f; ps2[r] = 0.f; }

#pragma unroll
    for (int t = 0; t < 4; ++t) {
        v8h out;
#pragma unroll
        for (int r = 0; r < 8; ++r) {
            const float v = accs[t][r] + bv[r];
            out[r] = (_Float16)v;
            ps1[r] += v;
            ps2[r] += v * v;
        }
        *(v8h*)(Y + (size_t)(colbase + t * 16 + n) * O + o0 + 8 * half) = out;
    }

    // butterfly reduce across the 16 lanes (same half) that share channel r
#pragma unroll
    for (int r = 0; r < 8; ++r) {
#pragma unroll
        for (int mxor = 1; mxor < 16; mxor <<= 1) {
            ps1[r] += __shfl_xor(ps1[r], mxor, 32);
            ps2[r] += __shfl_xor(ps2[r], mxor, 32);
        }
    }
    if (n == 0) {
#pragma unroll
        for (int r = 0; r < 8; ++r) {
            const int o = o0 + 8 * half + r;
            atomicAdd(&stats[o],     ps1[r]);
            atomicAdd(&stats[O + o], ps2[r]);
        }
    }
}

// ------------------------- BN apply / pooling -------------------------------

__global__ void zero_kernel(float* __restrict__ p, int n)
{
    int i = blockIdx.x * blockDim.x + threadIdx.x;
    if (i < n) p[i] = 0.f;
}

__global__ void apply_kernel(_Float16* __restrict__ Y,
                             const float* __restrict__ stats,
                             const float* __restrict__ g,
                             const float* __restrict__ beta,
                             const float* __restrict__ a,
                             int O, long long total, float inv_n)
{
    long long i = (long long)blockIdx.x * blockDim.x + threadIdx.x;
    const long long stride = (long long)gridDim.x * blockDim.x;
    for (; i < total; i += stride) {
        const int o = (int)(i % O);
        const float mu = stats[o] * inv_n;
        const float vv = stats[O + o] * inv_n - mu * mu;
        const float y  = (float)Y[i];
        const float yn = (y - mu) * rsqrtf(vv + EPS_) * g[o] + beta[o];
        Y[i] = (_Float16)(yn > 0.f ? yn : a[o] * yn);
    }
}

// Final layer: BN+PReLU, write f32 output. d_out layout (b,s,o) == col*256+o.
__global__ void apply_final_kernel(const _Float16* __restrict__ Y,
                                   const float* __restrict__ stats,
                                   const float* __restrict__ g,
                                   const float* __restrict__ beta,
                                   const float* __restrict__ a,
                                   float* __restrict__ out,
                                   int O, long long total, float inv_n)
{
    long long i = (long long)blockIdx.x * blockDim.x + threadIdx.x;
    const long long stride = (long long)gridDim.x * blockDim.x;
    for (; i < total; i += stride) {
        const int o = (int)(i % O);
        const float mu = stats[o] * inv_n;
        const float vv = stats[O + o] * inv_n - mu * mu;
        const float y  = (float)Y[i];
        const float yn = (y - mu) * rsqrtf(vv + EPS_) * g[o] + beta[o];
        out[i] = (yn > 0.f ? yn : a[o] * yn);
    }
}

// Max over K samples -> concat buffer (320 channels).
__global__ void maxpool_kernel(const _Float16* __restrict__ Yl,
                               _Float16* __restrict__ Xf,
                               int O, int K, int ch_off)
{
    const int idx = blockIdx.x * blockDim.x + threadIdx.x;
    if (idx >= Q_ * O) return;
    const int q = idx / O, o = idx - q * O;
    float mx = (float)Yl[(size_t)q * K * O + o];
    for (int k = 1; k < K; ++k) {
        const float v = (float)Yl[((size_t)q * K + k) * O + o];
        mx = fmaxf(mx, v);
    }
    Xf[(size_t)q * 320 + ch_off + o] = (_Float16)mx;
}

// f32 (O,C) weights -> f16 (O,Cpad) zero-padded.
__global__ void prep_w_kernel(const float* __restrict__ W,
                              _Float16* __restrict__ Wp,
                              int O, int C, int Cpad)
{
    const int i = blockIdx.x * blockDim.x + threadIdx.x;
    if (i >= O * Cpad) return;
    const int o = i / Cpad, c = i - o * Cpad;
    Wp[i] = (c < C) ? (_Float16)W[(size_t)o * C + c] : (_Float16)0.f;
}

// ------------------------------- host side ---------------------------------

extern "C" void kernel_launch(void* const* d_in, const int* in_sizes, int n_in,
                              void* d_out, int out_size, void* d_ws, size_t ws_size,
                              hipStream_t stream)
{
    (void)in_sizes; (void)n_in; (void)out_size; (void)ws_size;

    const float* new_xyz = (const float*)d_in[0];
    const float* xyz     = (const float*)d_in[1];
    const float* feat    = (const float*)d_in[2];

    // params leaves in jax pytree order: dict keys sorted -> 'blocks','final';
    // per-layer dict sorted -> 'W','a','b','beta','g'
    auto P = [&](int li, int which) -> const float* {
        return (const float*)d_in[3 + li * 5 + which];   // 0:W 1:a 2:b 3:beta 4:g
    };

    char* ws = (char*)d_ws;
    _Float16* bufA  = (_Float16*)(ws);
    _Float16* bufB  = (_Float16*)(ws + OFF_B);
    _Float16* xfin  = (_Float16*)(ws + OFF_XF);
    _Float16* wp    = (_Float16*)(ws + OFF_WP);
    float*    stats = (float*)(ws + OFF_ST);

    static const int    L_O [11] = {32,32,64,  64,64,128, 64,96,128, 256,256};
    static const int    L_C [11] = {67,32,32,  67,64,64,  67,64,96,  320,256};
    static const int    L_CP[11] = {96,32,32,  96,64,64,  96,64,96,  320,256};
    static const size_t L_WOFF[11] = {0,3072,4096,6144,12288,16384,24576,30720,36864,49152,131072};

    // pack weights to f16/padded
    for (int li = 0; li < 11; ++li) {
        const int n = L_O[li] * L_CP[li];
        prep_w_kernel<<<(n + 255) / 256, 256, 0, stream>>>(
            P(li, 0), wp + L_WOFF[li], L_O[li], L_C[li], L_CP[li]);
    }

    auto run_gemm = [&](int li, const _Float16* in, _Float16* out, long long cols) {
        const int O = L_O[li], Cpad = L_CP[li];
        zero_kernel<<<2, 256, 0, stream>>>(stats, 2 * O);
        wmma_gemm_kernel<<<dim3((unsigned)(cols / 256), O / 16), 128, 0, stream>>>(
            wp + L_WOFF[li], P(li, 2), in, out, stats, Cpad, O, cols);
    };

    auto run_layer = [&](int li, const _Float16* in, _Float16* out, long long cols) {
        const int O = L_O[li];
        run_gemm(li, in, out, cols);
        apply_kernel<<<2048, 256, 0, stream>>>(out, stats, P(li, 4), P(li, 3), P(li, 1),
                                               O, cols * O, 1.0f / (float)cols);
    };

    struct Blk { float r2; int K; long long cols; int li0; int ch_off; };
    const Blk blks[3] = {
        {0.01f, 16, (long long)Q_ * 16, 0, 0},
        {0.04f, 32, (long long)Q_ * 32, 3, 64},
        {0.16f, 64, (long long)Q_ * 64, 6, 192},
    };

    for (int bi = 0; bi < 3; ++bi) {
        const Blk& bk = blks[bi];
        group_kernel<<<Q_ / 8, 256, 0, stream>>>(new_xyz, xyz, feat, bufA, bk.r2, bk.K);
        run_layer(bk.li0 + 0, bufA, bufB, bk.cols);   // 67(96) -> O0
        run_layer(bk.li0 + 1, bufB, bufA, bk.cols);   // O0 -> O1
        run_layer(bk.li0 + 2, bufA, bufB, bk.cols);   // O1 -> O2
        const int O = L_O[bk.li0 + 2];
        maxpool_kernel<<<(Q_ * O + 255) / 256, 256, 0, stream>>>(bufB, xfin, O, bk.K, bk.ch_off);
    }

    // final MLP: 320 -> 256 -> 256, cols = B*S = 8192
    const long long fcols = Q_;
    run_layer(9, xfin, bufA, fcols);

    {   // layer 10: GEMM(+stats) -> BN/PReLU straight into d_out (f32)
        const int li = 10, O = 256;
        run_gemm(li, bufA, bufB, fcols);
        apply_final_kernel<<<2048, 256, 0, stream>>>(bufB, stats, P(li, 4), P(li, 3), P(li, 1),
                                                     (float*)d_out, O, fcols * O,
                                                     1.0f / (float)fcols);
    }
}
